// GPT2Model_11940009083087
// MI455X (gfx1250) — matrix operand order repaired and checked by reading, not compiled, not run on hardware
//
#include <hip/hip_runtime.h>
#include <math.h>

// ---------------- model constants ----------------
#define BQ 4           // batch
#define TT 1024        // seq len
#define DD 768         // model dim
#define HH 12          // heads
#define LL 12          // layers
#define DHD 64         // head dim
#define NTOK (BQ*TT)   // 4096 tokens
#define QKVLD (3*DD)   // 2304

// ---- gfx1250 async global->LDS path (guarded; falls back to reg staging) ----
#if defined(__has_builtin)
#  if __has_builtin(__builtin_amdgcn_global_load_async_to_lds_b128) && \
      __has_builtin(__builtin_amdgcn_s_wait_asynccnt)
#    define HAVE_ASYNC 1
#  endif
#endif
#ifndef HAVE_ASYNC
#  define HAVE_ASYNC 0
#endif

#if HAVE_ASYNC
typedef __attribute__((__vector_size__(4 * sizeof(int)))) int v4i_t;
__device__ __forceinline__ void async_cp16(const unsigned short* g, unsigned short* l) {
    __builtin_amdgcn_global_load_async_to_lds_b128((v4i_t*)g, (v4i_t*)l, 0, 0);
}
__device__ __forceinline__ void async_wait0() { __builtin_amdgcn_s_wait_asynccnt(0); }
#endif

typedef __attribute__((ext_vector_type(16))) __bf16 v16bf;
typedef __attribute__((ext_vector_type(8)))  float  v8f;

union Frag {
    v16bf v;
    uint4 d4[2];
    unsigned short h[16];
};

__device__ __forceinline__ unsigned short f2bf(float f) {
    unsigned int u = __builtin_bit_cast(unsigned int, f);
    unsigned int r = u + 0x7FFFu + ((u >> 16) & 1u);   // round-to-nearest-even
    return (unsigned short)(r >> 16);
}
__device__ __forceinline__ float gelu_f(float x) {
    const float c = 0.7978845608028654f;
    return 0.5f * x * (1.0f + tanhf(c * (x + 0.044715f * x * x * x)));
}

// ---------------- embedding: h = wte[ids] + wpe[pos] ----------------
__global__ __launch_bounds__(256) void embed_kernel(
    const int* __restrict__ ids, const float* __restrict__ wte,
    const float* __restrict__ wpe, float* __restrict__ h)
{
    const int row = blockIdx.x;              // token index 0..NTOK-1
    const int pos = row & (TT - 1);
    const size_t id = (size_t)ids[row];
    #pragma unroll
    for (int k = 0; k < 3; k++) {
        int i = threadIdx.x + k * 256;
        h[(size_t)row * DD + i] = wte[id * DD + i] + wpe[(size_t)pos * DD + i];
    }
}

// ---------------- f32 -> bf16 convert (4 elems / thread) ----------------
__global__ __launch_bounds__(256) void cvt_kernel(
    const float* __restrict__ in, unsigned short* __restrict__ out, int n)
{
    int i = (blockIdx.x * 256 + threadIdx.x) * 4;
    if (i < n) {
        float4 f = *(const float4*)(in + i);
        uint2 p;
        p.x = (unsigned int)f2bf(f.x) | ((unsigned int)f2bf(f.y) << 16);
        p.y = (unsigned int)f2bf(f.z) | ((unsigned int)f2bf(f.w) << 16);
        *(uint2*)(out + i) = p;
    }
}

// ---------------- layernorm over D=768, one row per block ----------------
template <bool BF16OUT>
__global__ __launch_bounds__(256) void layernorm_kernel(
    const float* __restrict__ x, const float* __restrict__ w,
    const float* __restrict__ b, float* outF, unsigned short* outB)
{
    __shared__ float r1[256];
    __shared__ float r2[256];
    const int tid = threadIdx.x;
    const size_t row = blockIdx.x;
    const float* xr = x + row * DD;
    float v0 = xr[tid], v1 = xr[tid + 256], v2 = xr[tid + 512];
    r1[tid] = v0 + v1 + v2;
    r2[tid] = v0 * v0 + v1 * v1 + v2 * v2;
    __syncthreads();
    for (int off = 128; off > 0; off >>= 1) {
        if (tid < off) { r1[tid] += r1[tid + off]; r2[tid] += r2[tid + off]; }
        __syncthreads();
    }
    const float mean = r1[0] * (1.0f / (float)DD);
    const float var  = r2[0] * (1.0f / (float)DD) - mean * mean;
    const float rstd = rsqrtf(var + 1e-5f);
    #pragma unroll
    for (int k = 0; k < 3; k++) {
        int i = tid + k * 256;
        float xv = (k == 0) ? v0 : (k == 1) ? v1 : v2;
        float y = (xv - mean) * rstd * w[i] + b[i];
        if (BF16OUT) outB[row * DD + i] = f2bf(y);
        else         outF[row * DD + i] = y;
    }
}

// ---------------- WMMA GEMM: C[M,N] = A[M,K](bf16) @ B[K,N](bf16) + bias ----------------
// block tile 128x128, k-step 32, 8 waves of 64x32 (4x2 wmma tiles).
// Double-buffered LDS: next tile staged (async-to-LDS when available) while
// WMMAs consume current -> single barrier per k-step.
template <bool GELU_ACT, bool RESIDUAL, bool OUT_BF16>
__global__ __launch_bounds__(256) void gemm_kernel(
    const unsigned short* __restrict__ A, const unsigned short* __restrict__ Bw,
    const float* __restrict__ bias, const float* res,
    float* outF, unsigned short* outB, int M, int N, int K)
{
    __shared__ unsigned short As[2][128 * 40];   // 128 rows x 32 k (+8 pad)
    __shared__ unsigned short Bs[2][32 * 136];   // 32 k x 128 n (+8 pad)
    const int tid   = threadIdx.x;
    const int lane  = tid & 31;
    const int wave  = tid >> 5;
    const int waveM = wave >> 2;              // 0..1
    const int waveN = wave & 3;               // 0..3
    const int l16   = lane & 15;
    const int hiH   = (lane >= 16);
    const int kOffA = hiH ? 8 : 0;
    const int m0 = blockIdx.y * 128;
    const int n0 = blockIdx.x * 128;

    const int arow = tid >> 1, ahalf = tid & 1;   // A tile: 2 threads per row
    const int brow = tid >> 3, bseg  = tid & 7;   // B tile: 8 threads per row

    const unsigned short* gA = A  + (size_t)(m0 + arow) * K + ahalf * 16;  // +k0
    const unsigned short* gB0 = Bw + (size_t)brow * N + n0 + bseg * 16;    // +k0*N
    unsigned short* lA0 = &As[0][arow * 40 + ahalf * 16];
    unsigned short* lB0 = &Bs[0][brow * 136 + bseg * 16];
    const int lbufA = 128 * 40, lbufB = 32 * 136;

    v8f acc[4][2];
    #pragma unroll
    for (int i = 0; i < 4; i++)
        #pragma unroll
        for (int j = 0; j < 2; j++) acc[i][j] = (v8f)0.0f;

    // ---- prologue: stage tile k0=0 into buffer 0 ----
#if HAVE_ASYNC
    async_cp16(gA, lA0);           async_cp16(gA + 8, lA0 + 8);
    async_cp16(gB0, lB0);          async_cp16(gB0 + 8, lB0 + 8);
    async_wait0();
#else
    {
        const uint4* sa = (const uint4*)gA;
        uint4 a0 = sa[0], a1 = sa[1];
        *(uint4*)(lA0) = a0; *(uint4*)(lA0 + 8) = a1;
        const uint4* sb = (const uint4*)gB0;
        uint4 b0 = sb[0], b1 = sb[1];
        *(uint4*)(lB0) = b0; *(uint4*)(lB0 + 8) = b1;
    }
#endif
    __syncthreads();

    int buf = 0;
    for (int k0 = 0; k0 < K; k0 += 32) {
        const bool more = (k0 + 32 < K);
        const int nb = buf ^ 1;
#if !HAVE_ASYNC
        uint4 ra0, ra1, rb0, rb1;
#endif
        if (more) {   // stage next tile into buf^1 (overlaps with compute)
            const unsigned short* ga = gA + (k0 + 32);
            const unsigned short* gb = gB0 + (size_t)(k0 + 32) * N;
#if HAVE_ASYNC
            unsigned short* la = lA0 + nb * lbufA;
            unsigned short* lb = lB0 + nb * lbufB;
            async_cp16(ga, la);  async_cp16(ga + 8, la + 8);
            async_cp16(gb, lb);  async_cp16(gb + 8, lb + 8);
#else
            const uint4* sa = (const uint4*)ga;
            ra0 = sa[0]; ra1 = sa[1];
            const uint4* sb = (const uint4*)gb;
            rb0 = sb[0]; rb1 = sb[1];
#endif
            if (k0 + 64 < K) {
                __builtin_prefetch(gA + (k0 + 64), 0, 0);
                __builtin_prefetch(gB0 + (size_t)(k0 + 64) * N, 0, 0);
            }
        }

        // ---- compute from current buffer ----
        Frag a[4];
        #pragma unroll
        for (int mt = 0; mt < 4; mt++) {
            const unsigned short* p = &As[buf][(waveM * 64 + mt * 16 + l16) * 40 + kOffA];
            a[mt].d4[0] = *(const uint4*)p;          // K 0..7 (lo) / 8..15 (hi)
            a[mt].d4[1] = *(const uint4*)(p + 16);   // K 16..23 (lo) / 24..31 (hi)
        }
        Frag bfr[2];
        #pragma unroll
        for (int nt = 0; nt < 2; nt++) {
            const unsigned short* p = &Bs[buf][waveN * 32 + nt * 16 + l16];
            #pragma unroll
            for (int e = 0; e < 16; e++)             // lane = N col, elems = K (table order)
                bfr[nt].h[e] = p[((e & 7) + ((e >> 3) << 4) + kOffA) * 136];
        }
        #pragma unroll
        for (int mt = 0; mt < 4; mt++)
            #pragma unroll
            for (int nt = 0; nt < 2; nt++)
                acc[mt][nt] = __builtin_amdgcn_wmma_f32_16x16x32_bf16(
                    false, a[mt].v, false, bfr[nt].v, (short)0, acc[mt][nt], false, false);

        if (more) {
#if HAVE_ASYNC
            async_wait0();
#else
            unsigned short* la = lA0 + nb * lbufA;
            unsigned short* lb = lB0 + nb * lbufB;
            *(uint4*)(la) = ra0; *(uint4*)(la + 8) = ra1;
            *(uint4*)(lb) = rb0; *(uint4*)(lb + 8) = rb1;
#endif
        }
        __syncthreads();
        buf = nb;
    }

    // epilogue: C layout -> VGPR j holds row (j + 8*hiH), lane l16 = col
    #pragma unroll
    for (int mt = 0; mt < 4; mt++) {
        #pragma unroll
        for (int nt = 0; nt < 2; nt++) {
            const int n = n0 + waveN * 32 + nt * 16 + l16;
            const float bv = bias[n];
            #pragma unroll
            for (int j = 0; j < 8; j++) {
                const int m = m0 + waveM * 64 + mt * 16 + j + (hiH ? 8 : 0);
                float v = acc[mt][nt][j] + bv;
                if (GELU_ACT) v = gelu_f(v);
                const size_t idx = (size_t)m * N + n;
                if (RESIDUAL) v += res[idx];
                if (OUT_BF16) outB[idx] = f2bf(v);
                else          outF[idx] = v;
            }
        }
    }
}

// ---------------- flash attention (S^T formulation, double-buffered K/V) ----------------
// 64 query rows / block, 4 waves x 16 query cols each.
// S^T = K @ Q^T: per-iteration A operands (K rows) are contiguous
// LDS b128 loads, V B-frags are strided gathers; loop-invariant Q^T is gathered once.  S^T in C-layout
// (lane=query, VGPR j=key) is bit-identical to the A-fragment layout of P for
// the P@V WMMA -> no LDS spill for P.  K/V tiles double-buffered.
__global__ __launch_bounds__(128) void attn_kernel(
    const unsigned short* __restrict__ qkv, unsigned short* __restrict__ out)
{
    __shared__ unsigned short Qs[64 * 72];
    __shared__ unsigned short Ks[2][64 * 72];
    __shared__ unsigned short Vs[2][64 * 72];
    const int tid  = threadIdx.x;
    const int lane = tid & 31;
    const int w    = tid >> 5;       // wave 0..3 -> queries w*16..+15
    const int l16  = lane & 15;
    const int hiH  = (lane >= 16);
    const int qb   = blockIdx.x;     // query block (64 rows)
    const int bb   = blockIdx.y / HH;
    const int hh   = blockIdx.y % HH;
    const int lrow = tid >> 1, lhalf = tid & 1;

    const unsigned short* gQ = qkv + ((size_t)bb * TT + (size_t)qb * 64 + lrow) * QKVLD
                                   + hh * DHD + lhalf * 32;
    const unsigned short* gK = qkv + ((size_t)bb * TT + lrow) * QKVLD
                                   + DD + hh * DHD + lhalf * 32;      // + kb*64*QKVLD
    const unsigned short* gV = gK + DD;
    unsigned short* lQ  = &Qs[lrow * 72 + lhalf * 32];
    unsigned short* lK0 = &Ks[0][lrow * 72 + lhalf * 32];
    unsigned short* lV0 = &Vs[0][lrow * 72 + lhalf * 32];
    const int lbuf = 64 * 72;

    // ---- prologue: stage Q tile and K/V tile kb=0 into buffer 0 ----
#if HAVE_ASYNC
    async_cp16(gQ, lQ);       async_cp16(gQ + 8,  lQ + 8);
    async_cp16(gQ + 16, lQ + 16); async_cp16(gQ + 24, lQ + 24);
    async_cp16(gK, lK0);      async_cp16(gK + 8,  lK0 + 8);
    async_cp16(gK + 16, lK0 + 16); async_cp16(gK + 24, lK0 + 24);
    async_cp16(gV, lV0);      async_cp16(gV + 8,  lV0 + 8);
    async_cp16(gV + 16, lV0 + 16); async_cp16(gV + 24, lV0 + 24);
    async_wait0();
#else
    {
        const uint4* s = (const uint4*)gQ;
        uint4 x0 = s[0], x1 = s[1], x2 = s[2], x3 = s[3];
        *(uint4*)(lQ) = x0; *(uint4*)(lQ + 8) = x1; *(uint4*)(lQ + 16) = x2; *(uint4*)(lQ + 24) = x3;
        const uint4* sk = (const uint4*)gK;
        uint4 k0 = sk[0], k1 = sk[1], k2 = sk[2], k3 = sk[3];
        *(uint4*)(lK0) = k0; *(uint4*)(lK0 + 8) = k1; *(uint4*)(lK0 + 16) = k2; *(uint4*)(lK0 + 24) = k3;
        const uint4* sv = (const uint4*)gV;
        uint4 v0 = sv[0], v1 = sv[1], v2 = sv[2], v3 = sv[3];
        *(uint4*)(lV0) = v0; *(uint4*)(lV0 + 8) = v1; *(uint4*)(lV0 + 16) = v2; *(uint4*)(lV0 + 24) = v3;
    }
#endif
    __syncthreads();

    // loop-invariant Q^T B-fragments: lane = query, elems = dh-in-slice (table order)
    Frag qt[2];
    #pragma unroll
    for (int ks = 0; ks < 2; ks++)
        #pragma unroll
        for (int e = 0; e < 16; e++)
            qt[ks].h[e] = Qs[(w * 16 + l16) * 72 + ks * 32 + (e & 7) + ((e >> 3) << 4) + (hiH ? 8 : 0)];

    // O accumulators: C = query x dh (lane = dh col, VGPR j = query row)
    v8f o[4];
    #pragma unroll
    for (int nt = 0; nt < 4; nt++) o[nt] = (v8f)0.0f;
    float m_l = -1e30f, l_l = 0.0f;          // per-lane softmax state (lane l16 = query)
    const int qcol = qb * 64 + w * 16 + l16; // this lane's query index

    int buf = 0;
    for (int kb = 0; kb <= qb; kb++) {
        const bool more = (kb < qb);
        const int nb = buf ^ 1;
#if !HAVE_ASYNC
        uint4 rk0, rk1, rk2, rk3, rv0, rv1, rv2, rv3;
#endif
        if (more) {   // stage K/V tile kb+1 into buf^1 (overlaps with compute)
            const unsigned short* gk = gK + (size_t)(kb + 1) * 64 * QKVLD;
            const unsigned short* gv = gk + DD;
#if HAVE_ASYNC
            unsigned short* lk = lK0 + nb * lbuf;
            unsigned short* lv = lV0 + nb * lbuf;
            async_cp16(gk, lk);       async_cp16(gk + 8,  lk + 8);
            async_cp16(gk + 16, lk + 16); async_cp16(gk + 24, lk + 24);
            async_cp16(gv, lv);       async_cp16(gv + 8,  lv + 8);
            async_cp16(gv + 16, lv + 16); async_cp16(gv + 24, lv + 24);
#else
            const uint4* sk = (const uint4*)gk;
            rk0 = sk[0]; rk1 = sk[1]; rk2 = sk[2]; rk3 = sk[3];
            const uint4* sv = (const uint4*)gv;
            rv0 = sv[0]; rv1 = sv[1]; rv2 = sv[2]; rv3 = sv[3];
#endif
        }

        // S^T = K @ Q^T : s[kt] covers keys kt*16..+15 x 16 queries
        v8f s[4];
        #pragma unroll
        for (int kt = 0; kt < 4; kt++) s[kt] = (v8f)0.0f;
        #pragma unroll
        for (int ks = 0; ks < 2; ks++) {      // dh slices of 32
            #pragma unroll
            for (int kt = 0; kt < 4; kt++) {
                Frag ak;  // A = K rows (contiguous in LDS)
                const unsigned short* p = &Ks[buf][(kt * 16 + l16) * 72 + ks * 32 + (hiH ? 8 : 0)];
                ak.d4[0] = *(const uint4*)p;
                ak.d4[1] = *(const uint4*)(p + 16);
                s[kt] = __builtin_amdgcn_wmma_f32_16x16x32_bf16(
                    false, ak.v, false, qt[ks].v, (short)0, s[kt], false, false);
            }
        }

        // scale + causal mask (element: key = VGPR row, query = lane)
        #pragma unroll
        for (int kt = 0; kt < 4; kt++)
            #pragma unroll
            for (int j = 0; j < 8; j++) {
                float v = s[kt][j] * 0.125f;
                const int key = kb * 64 + kt * 16 + j + (hiH ? 8 : 0);
                if (key > qcol) v = -1e30f;
                s[kt][j] = v;
            }

        // online softmax; query = lane, so only one cross-half reduce needed
        float rmax = -1e30f;
        #pragma unroll
        for (int kt = 0; kt < 4; kt++)
            #pragma unroll
            for (int j = 0; j < 8; j++) rmax = fmaxf(rmax, s[kt][j]);
        rmax = fmaxf(rmax, __shfl_xor(rmax, 16, 32));
        const float mnew  = fmaxf(m_l, rmax);
        const float alpha = __expf(m_l - mnew);
        float rs = 0.0f;
        #pragma unroll
        for (int kt = 0; kt < 4; kt++)
            #pragma unroll
            for (int j = 0; j < 8; j++) {
                float pv = __expf(s[kt][j] - mnew);
                s[kt][j] = pv;
                rs += pv;
            }
        rs += __shfl_xor(rs, 16, 32);
        l_l = l_l * alpha + rs;
        m_l = mnew;
        // rescale O rows: alpha is per-query -> broadcast from owning lane
        #pragma unroll
        for (int j = 0; j < 8; j++) {
            const float aj = __shfl(alpha, hiH ? (j + 8) : j, 32);
            #pragma unroll
            for (int nt = 0; nt < 4; nt++) o[nt][j] *= aj;
        }

        // P^T (C-layout) == A-fragment of P: pure in-register bf16 packing
        Frag pf[2];
        #pragma unroll
        for (int ks = 0; ks < 2; ks++) {
            #pragma unroll
            for (int e = 0; e < 8; e++) {
                pf[ks].h[e]     = f2bf(s[2 * ks + 0][e]);
                pf[ks].h[8 + e] = f2bf(s[2 * ks + 1][e]);
            }
        }

        // O += P @ V  (V B-frags: lane = dh col, elems = key-in-slice, table order)
        #pragma unroll
        for (int ks = 0; ks < 2; ks++) {
            #pragma unroll
            for (int nt = 0; nt < 4; nt++) {
                Frag bv;
                const unsigned short* q = &Vs[buf][(ks * 32) * 72 + nt * 16 + l16];
                #pragma unroll
                for (int e = 0; e < 16; e++)
                    bv.h[e] = q[((e & 7) + ((e >> 3) << 4) + (hiH ? 8 : 0)) * 72];
                o[nt] = __builtin_amdgcn_wmma_f32_16x16x32_bf16(
                    false, pf[ks].v, false, bv.v, (short)0, o[nt], false, false);
            }
        }

        if (more) {
#if HAVE_ASYNC
            async_wait0();
#else
            unsigned short* lk = lK0 + nb * lbuf;
            unsigned short* lv = lV0 + nb * lbuf;
            *(uint4*)(lk) = rk0; *(uint4*)(lk + 8) = rk1; *(uint4*)(lk + 16) = rk2; *(uint4*)(lk + 24) = rk3;
            *(uint4*)(lv) = rv0; *(uint4*)(lv + 8) = rv1; *(uint4*)(lv + 16) = rv2; *(uint4*)(lv + 24) = rv3;
#endif
        }
        __syncthreads();
        buf = nb;
    }

    // write O / l  (bf16, attention-output buffer laid out [token][D])
    #pragma unroll
    for (int j = 0; j < 8; j++) {
        const float lj = __shfl(l_l, hiH ? (j + 8) : j, 32);
        const float rinv = 1.0f / lj;
        const int row = w * 16 + j + (hiH ? 8 : 0);
        const size_t tok = (size_t)bb * TT + (size_t)qb * 64 + row;
        #pragma unroll
        for (int nt = 0; nt < 4; nt++)
            out[tok * DD + hh * DHD + nt * 16 + l16] = f2bf(o[nt][j] * rinv);
    }
}

// ---------------- host orchestration ----------------
static inline size_t alignup(size_t x) { return (x + 255) & ~(size_t)255; }

extern "C" void kernel_launch(void* const* d_in, const int* in_sizes, int n_in,
                              void* d_out, int out_size, void* d_ws, size_t ws_size,
                              hipStream_t stream) {
    const int*   ids    = (const int*)  d_in[0];
    const float* wte    = (const float*)d_in[1];
    const float* wpe    = (const float*)d_in[2];
    const float* ln1w   = (const float*)d_in[3];
    const float* ln1b   = (const float*)d_in[4];
    const float* attnw  = (const float*)d_in[5];
    const float* attnb  = (const float*)d_in[6];
    const float* attnpw = (const float*)d_in[7];
    const float* attnpb = (const float*)d_in[8];
    const float* ln2w   = (const float*)d_in[9];
    const float* ln2b   = (const float*)d_in[10];
    const float* fcw    = (const float*)d_in[11];
    const float* fcb    = (const float*)d_in[12];
    const float* projw  = (const float*)d_in[13];
    const float* projb  = (const float*)d_in[14];
    const float* lnfw   = (const float*)d_in[15];
    const float* lnfb   = (const float*)d_in[16];

    char* ws = (char*)d_ws;
    float*          h    = (float*)ws;          ws += alignup((size_t)NTOK * DD * 4);
    unsigned short* xb   = (unsigned short*)ws; ws += alignup((size_t)NTOK * DD * 2);
    unsigned short* qkvb = (unsigned short*)ws; ws += alignup((size_t)NTOK * 3 * DD * 2);
    unsigned short* midb = (unsigned short*)ws; ws += alignup((size_t)NTOK * 4 * DD * 2);
    unsigned short* wscr = (unsigned short*)ws; ws += alignup((size_t)DD * 4 * DD * 2);

    embed_kernel<<<NTOK, 256, 0, stream>>>(ids, wte, wpe, h);

    for (int l = 0; l < LL; l++) {
        // x = LN1(h) -> bf16
        layernorm_kernel<true><<<NTOK, 256, 0, stream>>>(h, ln1w + l * DD, ln1b + l * DD, nullptr, xb);
        // qkv = x @ attn_w + attn_b  -> bf16
        {
            int n = DD * 3 * DD;
            cvt_kernel<<<(n / 4 + 255) / 256, 256, 0, stream>>>(attnw + (size_t)l * n, wscr, n);
            gemm_kernel<false, false, true><<<dim3(3 * DD / 128, NTOK / 128), 256, 0, stream>>>(
                xb, wscr, attnb + (size_t)l * 3 * DD, nullptr, nullptr, qkvb, NTOK, 3 * DD, DD);
        }
        // a = flash_attention(q,k,v) -> bf16 into xb
        attn_kernel<<<dim3(TT / 64, BQ * HH), 128, 0, stream>>>(qkvb, xb);
        // h = h + a @ attnp_w + attnp_b
        {
            int n = DD * DD;
            cvt_kernel<<<(n / 4 + 255) / 256, 256, 0, stream>>>(attnpw + (size_t)l * n, wscr, n);
            gemm_kernel<false, true, false><<<dim3(DD / 128, NTOK / 128), 256, 0, stream>>>(
                xb, wscr, attnpb + (size_t)l * DD, h, h, nullptr, NTOK, DD, DD);
        }
        // x = LN2(h) -> bf16
        layernorm_kernel<true><<<NTOK, 256, 0, stream>>>(h, ln2w + l * DD, ln2b + l * DD, nullptr, xb);
        // m = gelu(x @ fc_w + fc_b) -> bf16
        {
            int n = DD * 4 * DD;
            cvt_kernel<<<(n / 4 + 255) / 256, 256, 0, stream>>>(fcw + (size_t)l * n, wscr, n);
            gemm_kernel<true, false, true><<<dim3(4 * DD / 128, NTOK / 128), 256, 0, stream>>>(
                xb, wscr, fcb + (size_t)l * 4 * DD, nullptr, nullptr, midb, NTOK, 4 * DD, DD);
        }
        // h = h + m @ proj_w + proj_b
        {
            int n = 4 * DD * DD;
            cvt_kernel<<<(n / 4 + 255) / 256, 256, 0, stream>>>(projw + (size_t)l * n, wscr, n);
            gemm_kernel<false, true, false><<<dim3(DD / 128, NTOK / 128), 256, 0, stream>>>(
                midb, wscr, projb + (size_t)l * DD, h, h, nullptr, NTOK, DD, 4 * DD);
        }
    }

    // out = LN_f(h) -> f32
    layernorm_kernel<false><<<NTOK, 256, 0, stream>>>(h, lnfw, lnfb, (float*)d_out, nullptr);
}
